// MemTransformerLM_10771777978445
// MI455X (gfx1250) — matrix operand order repaired and checked by reading, not compiled, not run on hardware
//
#include <hip/hip_runtime.h>
#include <hip/hip_bf16.h>
#include <stdint.h>

// ---- model dims (compile-time) ----
#define N_LAYER  4
#define N_HEADS  16
#define D_MODEL  1024
#define D_HEAD   64
#define D_INNER  4096
#define TGT_LEN  512
#define MEM_LEN  512
#define KLEN     1024          // MEM_LEN + TGT_LEN
#define BSZ      8
#define NTOK     (TGT_LEN * BSZ)   // 4096 query tokens

typedef __hip_bfloat16 bf16;
typedef __attribute__((ext_vector_type(16))) __bf16 v16bf;
typedef __attribute__((ext_vector_type(8)))  float  v8f;

union V16 { uint4 u[2]; v16bf v; };

// =====================================================================
// Generic batched bf16 WMMA GEMM:  C[z][m][n] = sum_k A[z][m][k]*B[z][n][k]
// Both A and B are K-contiguous (B holds rows of length K, i.e. B^T math).
// 4 waves / block in a 2x2 arrangement; each wave owns a 32x32 register
// tile (2x2 accumulators -> 4 WMMAs per 32-K step, 2 b128 loads per WMMA).
// Block tile = 64(M) x 64(N). All shapes in this model divide 64.
// Batch index z = z1*B2 + z2 with independent strides per tensor.
// Optional f32 and/or bf16 outputs, optional bias[n] and ReLU.
// =====================================================================
__global__ __launch_bounds__(128) void gemm_bf16_wmma(
    const bf16* __restrict__ A, const bf16* __restrict__ B,
    const float* __restrict__ bias,
    float* __restrict__ Cf, bf16* __restrict__ Cb,
    int K, int lda, int ldb, int ldc,
    long long aS1, long long aS2, long long bS1, long long bS2,
    long long cS1, long long cS2, int B2, int relu)
{
    const int z  = blockIdx.z;
    const int z1 = z / B2, z2 = z % B2;
    A += z1 * aS1 + z2 * aS2;
    B += z1 * bS1 + z2 * bS2;
    const long long coff = z1 * cS1 + z2 * cS2;

    const int lane = threadIdx.x & 31;
    const int wave = threadIdx.x >> 5;
    const int wm = wave >> 1;                       // m half within block
    const int wn = wave & 1;                        // n half within block
    const int m0 = (blockIdx.y << 6) + (wm << 5);   // wave tile: 32 rows
    const int n0 = (blockIdx.x << 6) + (wn << 5);   // wave tile: 32 cols
    const int r  = lane & 15;       // row (A) / col (B/C) within 16x16 tile
    const int kh = lane >> 4;       // K-half select per ISA 16-bit layouts

    // A lane view: elems 0..7 = K(kbase + kh*8 + 0..7), elems 8..15 = +16
    const bf16* ap0 = A + (long long)(m0 + r) * lda + (kh << 3);
    const bf16* ap1 = ap0 + (long long)16 * lda;
    // B lane view: elems 0..7 = K(kbase + kh*8 + 0..7), elems 8..15 = +16
    const bf16* bp0 = B + (long long)(n0 + r) * ldb + (kh << 3);
    const bf16* bp1 = bp0 + (long long)16 * ldb;

    v8f c00 = {}, c01 = {}, c10 = {}, c11 = {};
    for (int k0 = 0; k0 < K; k0 += 32) {
        V16 a0, a1, b0, b1;
        a0.u[0] = *(const uint4*)(ap0 + k0);
        a0.u[1] = *(const uint4*)(ap0 + k0 + 16);
        a1.u[0] = *(const uint4*)(ap1 + k0);
        a1.u[1] = *(const uint4*)(ap1 + k0 + 16);
        b0.u[0] = *(const uint4*)(bp0 + k0);
        b0.u[1] = *(const uint4*)(bp0 + k0 + 16);
        b1.u[0] = *(const uint4*)(bp1 + k0);
        b1.u[1] = *(const uint4*)(bp1 + k0 + 16);
        __builtin_prefetch(ap0 + k0 + 64, 0, 1);
        __builtin_prefetch(ap1 + k0 + 64, 0, 1);
        __builtin_prefetch(bp0 + k0 + 64, 0, 1);
        __builtin_prefetch(bp1 + k0 + 64, 0, 1);
        c00 = __builtin_amdgcn_wmma_f32_16x16x32_bf16(
                  false, a0.v, false, b0.v, (short)0, c00, false, false);
        c01 = __builtin_amdgcn_wmma_f32_16x16x32_bf16(
                  false, a0.v, false, b1.v, (short)0, c01, false, false);
        c10 = __builtin_amdgcn_wmma_f32_16x16x32_bf16(
                  false, a1.v, false, b0.v, (short)0, c10, false, false);
        c11 = __builtin_amdgcn_wmma_f32_16x16x32_bf16(
                  false, a1.v, false, b1.v, (short)0, c11, false, false);
    }

    v8f acc[2][2] = { { c00, c01 }, { c10, c11 } };
#pragma unroll
    for (int ni = 0; ni < 2; ++ni) {
        const int n = n0 + (ni << 4) + r;
        const float bi = bias ? bias[n] : 0.0f;
#pragma unroll
        for (int mi = 0; mi < 2; ++mi) {
#pragma unroll
            for (int i2 = 0; i2 < 8; ++i2) {
                const int m = m0 + (mi << 4) + i2 + (kh << 3);  // VGPR i -> M=i(+8)
                float v = acc[mi][ni][i2] + bi;
                if (relu) v = fmaxf(v, 0.0f);
                const long long idx = coff + (long long)m * ldc + n;
                if (Cf) Cf[idx] = v;
                if (Cb) Cb[idx] = __float2bfloat16(v);
            }
        }
    }
}

// =====================================================================
// Elementwise / prep kernels
// =====================================================================
__global__ void embed_kernel(const int* __restrict__ inp,
                             const float* __restrict__ tab,
                             float* __restrict__ core)
{
    const long long idx = (long long)blockIdx.x * blockDim.x + threadIdx.x;
    const int d = idx & (D_MODEL - 1);
    const int t = idx >> 10;                 // (q*BSZ + b)
    const int tok = inp[t];
    core[idx] = tab[(long long)tok * D_MODEL + d] * 32.0f;   // sqrt(1024)
}

__global__ void posemb_kernel(bf16* __restrict__ pe)
{
    const int idx = blockIdx.x * blockDim.x + threadIdx.x;   // KLEN * 512
    const int j = idx & 511;
    const int p = idx >> 9;
    const float pos  = (float)(KLEN - 1 - p);
    const float freq = __expf(-((float)(2 * j) / (float)D_MODEL) * 9.210340371976184f);
    const float a = pos * freq;
    pe[(long long)p * D_MODEL + j]       = __float2bfloat16(__sinf(a));
    pe[(long long)p * D_MODEL + 512 + j] = __float2bfloat16(__cosf(a));
}

__global__ void cvt_bf16_kernel(const float* __restrict__ in,
                                bf16* __restrict__ out, long long n)
{
    const long long i = (long long)blockIdx.x * blockDim.x + threadIdx.x;
    if (i < n) out[i] = __float2bfloat16(in[i]);
}

__global__ void prep_cat_kernel(const float* __restrict__ mems_l,
                                const float* __restrict__ core,
                                bf16* __restrict__ cat)
{
    const long long idx  = (long long)blockIdx.x * blockDim.x + threadIdx.x;
    const long long half = (long long)MEM_LEN * BSZ * D_MODEL;
    const float v = (idx < half) ? mems_l[idx] : core[idx - half];
    cat[idx] = __float2bfloat16(v);
}

// split heads (klen,b,3*D) -> qw/qr (b,h,i,d) with biases, kt (b,h,j,d), vt (b,h,d,j)
__global__ void prep_qkv_kernel(const bf16* __restrict__ heads,
                                const float* __restrict__ rwb,
                                const float* __restrict__ rrb,
                                bf16* __restrict__ qw, bf16* __restrict__ qr,
                                bf16* __restrict__ kt, bf16* __restrict__ vt)
{
    const long long idx = (long long)blockIdx.x * blockDim.x + threadIdx.x;
    const int d = idx & 63;
    const int h = (idx >> 6) & 15;
    const int b = (idx >> 10) & 7;
    const int j = idx >> 13;
    const long long hb = ((long long)j * BSZ + b) * (3 * D_MODEL);
    const int hd = h * D_HEAD + d;
    const long long bh = (long long)(b * N_HEADS + h);
    kt[(bh * KLEN + j) * D_HEAD + d]  = heads[hb + D_MODEL + hd];
    vt[(bh * D_HEAD + d) * KLEN + j]  = heads[hb + 2 * D_MODEL + hd];
    if (j >= MEM_LEN) {
        const int i = j - MEM_LEN;
        const float qv = __bfloat162float(heads[hb + hd]);
        qw[(bh * TGT_LEN + i) * D_HEAD + d] = __float2bfloat16(qv + rwb[hd]);
        qr[(bh * TGT_LEN + i) * D_HEAD + d] = __float2bfloat16(qv + rrb[hd]);
    }
}

// rk (j, H*dh) -> rk_t (h, j, d)
__global__ void prep_rk_kernel(const bf16* __restrict__ rk, bf16* __restrict__ rkt)
{
    const long long idx = (long long)blockIdx.x * blockDim.x + threadIdx.x;
    const int d = idx & 63;
    const int j = (idx >> 6) & 1023;
    const int h = idx >> 16;
    rkt[((long long)h * KLEN + j) * D_HEAD + d] =
        rk[(long long)j * D_MODEL + h * D_HEAD + d];
}

// scores = (AC + relshift(BD)) * scale, causal+mem mask, softmax over klen
__global__ __launch_bounds__(128) void score_softmax_kernel(
    const float* __restrict__ AC, const float* __restrict__ BD,
    bf16* __restrict__ prob)
{
    const int i = blockIdx.x, h = blockIdx.y, b = blockIdx.z;
    const long long bh = (long long)(b * N_HEADS + h);
    const float* ac = AC + (bh * TGT_LEN + i) * KLEN;
    const float* bd = BD + (bh * TGT_LEN + i) * KLEN;
    bf16* pr = prob + (bh * TGT_LEN + i) * KLEN;
    const float scale = 0.125f;   // 1/sqrt(64)

    float s[8];
    float mx = -3.4e38f;
#pragma unroll
    for (int t = 0; t < 8; ++t) {
        const int j = threadIdx.x + (t << 7);
        float v;
        if (j > i + MEM_LEN) {
            v = -1e30f;
        } else {
            // rel_shift: BD_shifted[i,j] = BD_raw[i, j + qlen-1 - i]
            const int jr = j + (TGT_LEN - 1) - i;
            const float bdv = (jr < KLEN) ? bd[jr] : 0.0f;
            v = (ac[j] + bdv) * scale;
        }
        s[t] = v; mx = fmaxf(mx, v);
    }
    __shared__ float red[128];
    red[threadIdx.x] = mx; __syncthreads();
    for (int o = 64; o > 0; o >>= 1) {
        if (threadIdx.x < o) red[threadIdx.x] = fmaxf(red[threadIdx.x], red[threadIdx.x + o]);
        __syncthreads();
    }
    mx = red[0]; __syncthreads();
    float sum = 0.0f;
#pragma unroll
    for (int t = 0; t < 8; ++t) { s[t] = __expf(s[t] - mx); sum += s[t]; }
    red[threadIdx.x] = sum; __syncthreads();
    for (int o = 64; o > 0; o >>= 1) {
        if (threadIdx.x < o) red[threadIdx.x] += red[threadIdx.x + o];
        __syncthreads();
    }
    const float inv = 1.0f / red[0];
#pragma unroll
    for (int t = 0; t < 8; ++t) {
        const int j = threadIdx.x + (t << 7);
        pr[j] = __float2bfloat16(s[t] * inv);
    }
}

// out = LayerNorm(x + delta) * g + b ; writes f32 + bf16 copies
__global__ __launch_bounds__(256) void add_layernorm_kernel(
    const float* __restrict__ x, const float* __restrict__ dlt,
    const float* __restrict__ g, const float* __restrict__ bta,
    float* __restrict__ outf, bf16* __restrict__ outb)
{
    const long long base = (long long)blockIdx.x * D_MODEL;
    float v[4];
#pragma unroll
    for (int t = 0; t < 4; ++t) {
        const int d = threadIdx.x + (t << 8);
        v[t] = x[base + d] + dlt[base + d];
    }
    __shared__ float red[256];
    red[threadIdx.x] = v[0] + v[1] + v[2] + v[3]; __syncthreads();
    for (int o = 128; o > 0; o >>= 1) {
        if (threadIdx.x < o) red[threadIdx.x] += red[threadIdx.x + o];
        __syncthreads();
    }
    const float mean = red[0] * (1.0f / D_MODEL); __syncthreads();
    float vs = 0.0f;
#pragma unroll
    for (int t = 0; t < 4; ++t) { const float d0 = v[t] - mean; vs += d0 * d0; }
    red[threadIdx.x] = vs; __syncthreads();
    for (int o = 128; o > 0; o >>= 1) {
        if (threadIdx.x < o) red[threadIdx.x] += red[threadIdx.x + o];
        __syncthreads();
    }
    const float rstd = rsqrtf(red[0] * (1.0f / D_MODEL) + 1e-5f);
#pragma unroll
    for (int t = 0; t < 4; ++t) {
        const int d = threadIdx.x + (t << 8);
        const float o2 = (v[t] - mean) * rstd * g[d] + bta[d];
        outf[base + d] = o2;
        outb[base + d] = __float2bfloat16(o2);
    }
}

// =====================================================================
// Host-side orchestration
// =====================================================================
extern "C" void kernel_launch(void* const* d_in, const int* in_sizes, int n_in,
                              void* d_out, int out_size, void* d_ws, size_t ws_size,
                              hipStream_t stream) {
    (void)in_sizes; (void)n_in; (void)ws_size;
    const int*   inp    = (const int*)  d_in[0];
    const float* mems   = (const float*)d_in[1];
    const float* emb    = (const float*)d_in[2];
    const float* rwb    = (const float*)d_in[3];
    const float* rrb    = (const float*)d_in[4];
    const float* qkv_w  = (const float*)d_in[5];
    const float* r_w    = (const float*)d_in[6];
    const float* o_w    = (const float*)d_in[7];
    const float* ln1_g  = (const float*)d_in[8];
    const float* ln1_b  = (const float*)d_in[9];
    const float* ff_w1  = (const float*)d_in[10];
    const float* ff_b1  = (const float*)d_in[11];
    const float* ff_w2  = (const float*)d_in[12];
    const float* ff_b2  = (const float*)d_in[13];
    const float* ln2_g  = (const float*)d_in[14];
    const float* ln2_b  = (const float*)d_in[15];

    // ---- workspace partition ----
    char* p = (char*)d_ws;
    auto alloc = [&](size_t bytes) -> char* {
        char* r = p; p += (bytes + 255) & ~(size_t)255; return r;
    };
    const long long BHQ = (long long)BSZ * N_HEADS * TGT_LEN * KLEN;  // 67.1M
    float* core    = (float*)alloc((size_t)NTOK * D_MODEL * 4);
    float* delta   = (float*)alloc((size_t)NTOK * D_MODEL * 4);
    float* AC      = (float*)alloc((size_t)BHQ * 4);
    float* BD      = (float*)alloc((size_t)BHQ * 4);
    bf16* core_bf  = (bf16*)alloc((size_t)NTOK * D_MODEL * 2);
    bf16* cat_bf   = (bf16*)alloc((size_t)KLEN * BSZ * D_MODEL * 2);
    bf16* heads_bf = (bf16*)alloc((size_t)KLEN * BSZ * 3 * D_MODEL * 2);
    bf16* pe_bf    = (bf16*)alloc((size_t)KLEN * D_MODEL * 2);
    bf16* rk_bf    = (bf16*)alloc((size_t)KLEN * D_MODEL * 2);
    bf16* rkt_bf   = (bf16*)alloc((size_t)N_HEADS * KLEN * D_HEAD * 2);
    bf16* qw_bf    = (bf16*)alloc((size_t)BSZ * N_HEADS * TGT_LEN * D_HEAD * 2);
    bf16* qr_bf    = (bf16*)alloc((size_t)BSZ * N_HEADS * TGT_LEN * D_HEAD * 2);
    bf16* kt_bf    = (bf16*)alloc((size_t)BSZ * N_HEADS * KLEN * D_HEAD * 2);
    bf16* vt_bf    = (bf16*)alloc((size_t)BSZ * N_HEADS * D_HEAD * KLEN * 2);
    bf16* prob_bf  = (bf16*)alloc((size_t)BHQ * 2);
    bf16* av_bf    = (bf16*)alloc((size_t)NTOK * D_MODEL * 2);
    bf16* h1_bf    = (bf16*)alloc((size_t)NTOK * D_INNER * 2);
    bf16* qkvw_bf  = (bf16*)alloc((size_t)N_LAYER * 3 * D_MODEL * D_MODEL * 2);
    bf16* rw_bf    = (bf16*)alloc((size_t)N_LAYER * D_MODEL * D_MODEL * 2);
    bf16* ow_bf    = (bf16*)alloc((size_t)N_LAYER * D_MODEL * D_MODEL * 2);
    bf16* fw1_bf   = (bf16*)alloc((size_t)N_LAYER * D_INNER * D_MODEL * 2);
    bf16* fw2_bf   = (bf16*)alloc((size_t)N_LAYER * D_MODEL * D_INNER * 2);

    auto cvt = [&](const float* in, bf16* out, long long n) {
        hipLaunchKernelGGL(cvt_bf16_kernel, dim3((unsigned)((n + 255) / 256)),
                           dim3(256), 0, stream, in, out, n);
    };
    auto gemm = [&](const bf16* A, const bf16* B, const float* bias,
                    float* Cf, bf16* Cb, int M, int N, int K,
                    int lda, int ldb, int ldc,
                    long long aS1, long long aS2, long long bS1, long long bS2,
                    long long cS1, long long cS2, int B1, int B2, int relu) {
        dim3 grid((unsigned)(N / 64), (unsigned)(M / 64), (unsigned)(B1 * B2));
        hipLaunchKernelGGL(gemm_bf16_wmma, grid, dim3(128), 0, stream,
                           A, B, bias, Cf, Cb, K, lda, ldb, ldc,
                           aS1, aS2, bS1, bS2, cS1, cS2, B2, relu);
    };

    // ---- one-time (per call) conversions / precompute ----
    cvt(qkv_w, qkvw_bf, (long long)N_LAYER * 3 * D_MODEL * D_MODEL);
    cvt(r_w,   rw_bf,   (long long)N_LAYER * D_MODEL * D_MODEL);
    cvt(o_w,   ow_bf,   (long long)N_LAYER * D_MODEL * D_MODEL);
    cvt(ff_w1, fw1_bf,  (long long)N_LAYER * D_INNER * D_MODEL);
    cvt(ff_w2, fw2_bf,  (long long)N_LAYER * D_MODEL * D_INNER);

    hipLaunchKernelGGL(embed_kernel, dim3(NTOK * D_MODEL / 256), dim3(256), 0, stream,
                       inp, emb, core);
    hipLaunchKernelGGL(posemb_kernel, dim3(KLEN * 512 / 256), dim3(256), 0, stream, pe_bf);

    const long long qS2 = (long long)TGT_LEN * D_HEAD;           // per-head q stride
    const long long qS1 = (long long)N_HEADS * qS2;              // per-batch
    const long long kS2 = (long long)KLEN * D_HEAD;
    const long long kS1 = (long long)N_HEADS * kS2;
    const long long sS2 = (long long)TGT_LEN * KLEN;             // score per-head
    const long long sS1 = (long long)N_HEADS * sS2;

    for (int l = 0; l < N_LAYER; ++l) {
        // cat = [mems[l]; core]  (bf16)
        hipLaunchKernelGGL(prep_cat_kernel, dim3(KLEN * BSZ * D_MODEL / 256), dim3(256),
                           0, stream, mems + (long long)l * MEM_LEN * BSZ * D_MODEL,
                           core, cat_bf);
        // heads = cat @ qkv_w^T  : (8192 x 3072), K=1024
        gemm(cat_bf, qkvw_bf + (long long)l * 3 * D_MODEL * D_MODEL, nullptr,
             nullptr, heads_bf, KLEN * BSZ, 3 * D_MODEL, D_MODEL,
             D_MODEL, D_MODEL, 3 * D_MODEL, 0, 0, 0, 0, 0, 0, 1, 1, 0);
        hipLaunchKernelGGL(prep_qkv_kernel,
                           dim3(KLEN * BSZ * N_HEADS * D_HEAD / 256), dim3(256), 0, stream,
                           heads_bf, rwb, rrb, qw_bf, qr_bf, kt_bf, vt_bf);
        // rk = pos_emb @ r_w^T : (1024 x 1024), then (h,j,d)
        gemm(pe_bf, rw_bf + (long long)l * D_MODEL * D_MODEL, nullptr,
             nullptr, rk_bf, KLEN, D_MODEL, D_MODEL,
             D_MODEL, D_MODEL, D_MODEL, 0, 0, 0, 0, 0, 0, 1, 1, 0);
        hipLaunchKernelGGL(prep_rk_kernel, dim3(N_HEADS * KLEN * D_HEAD / 256),
                           dim3(256), 0, stream, rk_bf, rkt_bf);
        // AC[b,h] = qw @ kt^T : (512 x 1024), K=64, batch 8x16
        gemm(qw_bf, kt_bf, nullptr, AC, nullptr, TGT_LEN, KLEN, D_HEAD,
             D_HEAD, D_HEAD, KLEN, qS1, qS2, kS1, kS2, sS1, sS2, BSZ, N_HEADS, 0);
        // BD_raw[b,h] = qr @ rkt[h]^T : same shape; B batched over h only
        gemm(qr_bf, rkt_bf, nullptr, BD, nullptr, TGT_LEN, KLEN, D_HEAD,
             D_HEAD, D_HEAD, KLEN, qS1, qS2, 0, kS2, sS1, sS2, BSZ, N_HEADS, 0);
        // mask + rel-shift + softmax -> prob (bf16)
        hipLaunchKernelGGL(score_softmax_kernel, dim3(TGT_LEN, N_HEADS, BSZ),
                           dim3(128), 0, stream, AC, BD, prob_bf);
        // attn_vec[b,h] = prob @ vt^T : (512 x 64), K=1024; C scattered to (i,b,h,d)
        gemm(prob_bf, vt_bf, nullptr, nullptr, av_bf, TGT_LEN, D_HEAD, KLEN,
             KLEN, KLEN, BSZ * D_MODEL,
             sS1, sS2, (long long)N_HEADS * D_HEAD * KLEN, (long long)D_HEAD * KLEN,
             (long long)D_MODEL, (long long)D_HEAD, BSZ, N_HEADS, 0);
        // attn_out = attn_vec @ o_w^T : (4096 x 1024), K=1024
        gemm(av_bf, ow_bf + (long long)l * D_MODEL * D_MODEL, nullptr,
             delta, nullptr, NTOK, D_MODEL, D_MODEL,
             D_MODEL, D_MODEL, D_MODEL, 0, 0, 0, 0, 0, 0, 1, 1, 0);
        // core = LN1(core + attn_out)
        hipLaunchKernelGGL(add_layernorm_kernel, dim3(NTOK), dim3(256), 0, stream,
                           core, delta, ln1_g + l * D_MODEL, ln1_b + l * D_MODEL,
                           core, core_bf);
        // h1 = relu(core @ ff_w1^T + b1) : (4096 x 4096), K=1024
        gemm(core_bf, fw1_bf + (long long)l * D_INNER * D_MODEL, ff_b1 + l * D_INNER,
             nullptr, h1_bf, NTOK, D_INNER, D_MODEL,
             D_MODEL, D_MODEL, D_INNER, 0, 0, 0, 0, 0, 0, 1, 1, 1);
        // ff = h1 @ ff_w2^T + b2 : (4096 x 1024), K=4096
        gemm(h1_bf, fw2_bf + (long long)l * D_MODEL * D_INNER, ff_b2 + l * D_MODEL,
             delta, nullptr, NTOK, D_MODEL, D_INNER,
             D_INNER, D_INNER, D_MODEL, 0, 0, 0, 0, 0, 0, 1, 1, 0);
        // core = LN2(core + ff)
        hipLaunchKernelGGL(add_layernorm_kernel, dim3(NTOK), dim3(256), 0, stream,
                           core, delta, ln2_g + l * D_MODEL, ln2_b + l * D_MODEL,
                           core, core_bf);
    }

    hipMemcpyAsync(d_out, core, (size_t)out_size * sizeof(float),
                   hipMemcpyDeviceToDevice, stream);
}
